// RAUCell_42683384987795
// MI455X (gfx1250) — matrix operand order repaired and checked by reading, not compiled, not run on hardware
//
#include <hip/hip_runtime.h>
#include <hip/hip_bf16.h>
#include <math.h>

// ---------------------------------------------------------------------------
// RAU cell, MI455X (gfx1250, wave32) — bf16 WMMA GEMMs with fused epilogues,
// double-buffered LDS staging via CDNA5 async global->LDS copies (ASYNCcnt).
//   GEMM1: combined[8192,2048] @ W4[4096,2048]^T  -> c/hat/rh/z (fused epi)
//   GEMM2: x[8192,1024]        @ Wxh[1024,1024]^T -> xh (+bias_h)
//   GEMM3: rh[8192,1024]       @ Whh[1024,1024]^T -> xh += (accumulate epi)
//   softmax(c) rows, then final elementwise combine.
// ---------------------------------------------------------------------------

typedef __bf16 bf16_t;
typedef __attribute__((ext_vector_type(16))) __bf16 v16bf;
typedef __attribute__((ext_vector_type(8)))  __bf16 v8bf;
typedef __attribute__((ext_vector_type(8)))  float  v8f;
typedef __attribute__((ext_vector_type(4)))  int    v4i;

#define HID 1024
#define BATCH 8192
#define KCOMB 2048   // I + H

#if __has_builtin(__builtin_amdgcn_global_load_async_to_lds_b128) && \
    __has_builtin(__builtin_amdgcn_s_wait_asynccnt)
#define USE_ASYNC_LDS 1
#else
#define USE_ASYNC_LDS 0
#endif

__device__ __forceinline__ float sigmoidf_(float x) {
    return 1.0f / (1.0f + expf(-x));
}

#if USE_ASYNC_LDS
// 16-byte async copy global -> LDS (GLOBAL_LOAD_ASYNC_TO_LDS_B128).
// Builtin signature (from compiler diagnostic): param0 is AS1 pointer to
// int __attribute__((vector_size(16))); LDS side is AS3.
__device__ __forceinline__ void async_copy16(const bf16_t* g, bf16_t* l) {
    __builtin_amdgcn_global_load_async_to_lds_b128(
        (__attribute__((address_space(1))) v4i*)const_cast<bf16_t*>(g),
        (__attribute__((address_space(3))) v4i*)l,
        0, 0);
}
#endif

// ------------------------- conversion / packing ----------------------------

__global__ __launch_bounds__(256)
void k_build_combined(const float* __restrict__ x, const float* __restrict__ h,
                      bf16_t* __restrict__ comb) {
    size_t i = (size_t)blockIdx.x * 256 + threadIdx.x;   // over BATCH*HID
    if (i >= (size_t)BATCH * HID) return;
    size_t m = i / HID, n = i % HID;
    comb[m * KCOMB + n]       = (bf16_t)x[i];
    comb[m * KCOMB + HID + n] = (bf16_t)h[i];
}

__global__ __launch_bounds__(256)
void k_pack_w4(const float* __restrict__ wc,  const float* __restrict__ what,
               const float* __restrict__ wxr, const float* __restrict__ whr,
               const float* __restrict__ wxz, const float* __restrict__ whz,
               bf16_t* __restrict__ W4) {
    size_t i = (size_t)blockIdx.x * 256 + threadIdx.x;   // over 4096*2048
    if (i >= (size_t)4 * HID * KCOMB) return;
    size_t row = i / KCOMB, col = i % KCOMB;
    float v;
    if (row < HID)            v = wc[row * KCOMB + col];
    else if (row < 2 * HID)   v = what[(row - HID) * KCOMB + col];
    else if (row < 3 * HID) { size_t n = row - 2 * HID;
        v = (col < HID) ? wxr[n * HID + col] : whr[n * HID + col - HID]; }
    else                    { size_t n = row - 3 * HID;
        v = (col < HID) ? wxz[n * HID + col] : whz[n * HID + col - HID]; }
    W4[i] = (bf16_t)v;
}

__global__ __launch_bounds__(256)
void k_cvt(const float* __restrict__ src, bf16_t* __restrict__ dst, size_t n) {
    size_t i = (size_t)blockIdx.x * 256 + threadIdx.x;
    if (i < n) dst[i] = (bf16_t)src[i];
}

// ------------------------------ epilogues ----------------------------------

struct EpiFused1 {   // GEMM1: N-region 0:c  1:hat(relu)  2:r->rh(bf16)  3:z
    float* c; float* hat; bf16_t* rh; float* z;
    const float* bc; const float* bhat; const float* br; const float* bz;
    const float* hid;
    __device__ void store(int m, int n, float v) const {
        int region = n >> 10, nn = n & (HID - 1);
        size_t idx = (size_t)m * HID + nn;
        if (region == 0)      c[idx] = v + bc[nn];
        else if (region == 1) hat[idx] = fmaxf(v + bhat[nn], 0.0f);
        else if (region == 2) { float r = sigmoidf_(v + br[nn]);
                                rh[idx] = (bf16_t)(r * hid[idx]); }
        else                  z[idx] = sigmoidf_(v + bz[nn]);
    }
};

struct EpiBias {     // out = acc + bias[n]
    float* out; const float* bias;
    __device__ void store(int m, int n, float v) const {
        out[(size_t)m * HID + n] = v + bias[n];
    }
};

struct EpiAcc {      // out += acc   (GEMM3 accumulates onto xh buffer)
    float* out;
    __device__ void store(int m, int n, float v) const {
        out[(size_t)m * HID + n] += v;
    }
};

// ------------------------- WMMA bf16 GEMM core -----------------------------
// C[M,N] = A[M,K](row-major, stride lda) @ W[N,K](row-major)^T, epilogue epi.
// Block tile 128x128, BK=64; 8 waves (2x4), each wave 64x32 = 4x2 fragments.
// Double-buffered LDS; async global->LDS copy when available.

template <class Epi>
__global__ __launch_bounds__(256)
void gemm_bf16_wmma(const bf16_t* __restrict__ Ag, int lda,
                    const bf16_t* __restrict__ Wg, int K, Epi epi) {
    constexpr int BM = 128, BN = 128, BK = 64;
    __shared__ bf16_t As[2][BM * BK];
    __shared__ bf16_t Bs[2][BN * BK];

    const int tid    = threadIdx.x;
    const int lane   = tid & 31;
    const int wid    = tid >> 5;
    const int wave_m = wid >> 2;         // 0..1
    const int wave_n = wid & 3;          // 0..3
    const int bm     = blockIdx.y * BM;
    const int bn     = blockIdx.x * BN;
    const int l      = lane & 15;        // fragment row/col within half-wave
    const int hi     = lane >> 4;        // half-wave select

    v8f acc[4][2];
#pragma unroll
    for (int i = 0; i < 4; ++i)
#pragma unroll
        for (int j = 0; j < 2; ++j)
#pragma unroll
            for (int r = 0; r < 8; ++r) acc[i][j][r] = 0.0f;

    // One k-tile of WMMA work on LDS buffer p.
    auto compute = [&](int p) {
#pragma unroll
        for (int kk = 0; kk < BK; kk += 32) {
            // B fragment (32x16 KxN): lanes 0-15 hold K 0..15, 16-31 hold 16..31
            v16bf bfrag[2];
#pragma unroll
            for (int j = 0; j < 2; ++j)
                bfrag[j] = *(const v16bf*)
                    &Bs[p][(wave_n * 32 + j * 16 + l) * BK + kk + hi * 16];

#pragma unroll
            for (int i = 0; i < 4; ++i) {
                // A fragment (16x32 MxK): lanes 0-15 K 0..15,
                //                         lanes 16-31 K 16..31
                const bf16_t* pa =
                    &As[p][(wave_m * 64 + i * 16 + l) * BK + kk + hi * 16];
                v8bf alo = *(const v8bf*)pa;
                v8bf ahi = *(const v8bf*)(pa + 8);
                v16bf afrag;
#pragma unroll
                for (int e = 0; e < 8; ++e) { afrag[e] = alo[e]; afrag[e + 8] = ahi[e]; }

#pragma unroll
                for (int j = 0; j < 2; ++j)
                    acc[i][j] = __builtin_amdgcn_wmma_f32_16x16x32_bf16(
                        false, afrag, false, bfrag[j], (short)0, acc[i][j],
                        false, false);
            }
        }
    };

#if USE_ASYNC_LDS
    // Fire-and-forget async staging of one 128x64 A tile + 128x64 B tile
    // (1024 16B chunks, 4 per thread per matrix) into buffer p.
    auto stage = [&](int k0, int p) {
#pragma unroll
        for (int i = 0; i < 4; ++i) {
            int ch  = tid + i * 256;          // 0..1023
            int row = ch >> 3;
            int cc  = (ch & 7) * 8;
            async_copy16(&Ag[(size_t)(bm + row) * lda + k0 + cc],
                         &As[p][row * BK + cc]);
            async_copy16(&Wg[(size_t)(bn + row) * K + k0 + cc],
                         &Bs[p][row * BK + cc]);
        }
    };

    stage(0, 0);
    __builtin_amdgcn_s_wait_asynccnt(0);
    __syncthreads();
    int p = 0;
    for (int k0 = 0; k0 < K; k0 += BK) {
        if (k0 + BK < K) stage(k0 + BK, p ^ 1);   // overlap copy(t+1) w/ wmma(t)
        compute(p);
        __builtin_amdgcn_s_wait_asynccnt(0);
        __syncthreads();
        p ^= 1;
    }
#else
    // Synchronous fallback: load->VGPR->LDS staging, single buffer.
    for (int k0 = 0; k0 < K; k0 += BK) {
#pragma unroll
        for (int i = 0; i < 4; ++i) {
            int ch  = tid + i * 256;
            int row = ch >> 3;
            int cc  = (ch & 7) * 8;
            *(v8bf*)&As[0][row * BK + cc] =
                *(const v8bf*)&Ag[(size_t)(bm + row) * lda + k0 + cc];
            *(v8bf*)&Bs[0][row * BK + cc] =
                *(const v8bf*)&Wg[(size_t)(bn + row) * K + k0 + cc];
        }
        __syncthreads();
        compute(0);
        __syncthreads();
    }
#endif

    // C/D layout: VGPR r -> M = 8*hi + r (within 16), N = l (within 16).
#pragma unroll
    for (int i = 0; i < 4; ++i)
#pragma unroll
        for (int j = 0; j < 2; ++j) {
            int mbase = bm + wave_m * 64 + i * 16 + hi * 8;
            int n     = bn + wave_n * 32 + j * 16 + l;
#pragma unroll
            for (int r = 0; r < 8; ++r) epi.store(mbase + r, n, acc[i][j][r]);
        }
}

// --------------------------- softmax / final -------------------------------

__global__ __launch_bounds__(256)
void softmax_rows(float* __restrict__ c) {
    __shared__ float red[256];
    float* p = c + (size_t)blockIdx.x * HID;
    int t = threadIdx.x;
    float mx = -INFINITY;
    for (int i = t; i < HID; i += 256) mx = fmaxf(mx, p[i]);
    red[t] = mx; __syncthreads();
    for (int s = 128; s > 0; s >>= 1) {
        if (t < s) red[t] = fmaxf(red[t], red[t + s]);
        __syncthreads();
    }
    mx = red[0]; __syncthreads();
    float sum = 0.0f;
    for (int i = t; i < HID; i += 256) { float e = expf(p[i] - mx); p[i] = e; sum += e; }
    red[t] = sum; __syncthreads();
    for (int s = 128; s > 0; s >>= 1) {
        if (t < s) red[t] += red[t + s];
        __syncthreads();
    }
    float inv = 1.0f / red[0];
    for (int i = t; i < HID; i += 256) p[i] *= inv;
}

__global__ __launch_bounds__(256)
void k_final(const float* __restrict__ z, const float* __restrict__ xh_hh,
             const float* __restrict__ hid, const float* __restrict__ a,
             const float* __restrict__ hat, float* __restrict__ out) {
    size_t i = (size_t)blockIdx.x * 256 + threadIdx.x;
    if (i >= (size_t)BATCH * HID) return;
    float zz = z[i];
    float ht = tanhf(xh_hh[i]);                        // xh + hh (+bias_h)
    out[i] = (1.0f - zz) * ht + zz * hid[i] + a[i] * hat[i];
}

// ------------------------------- launcher ----------------------------------

extern "C" void kernel_launch(void* const* d_in, const int* in_sizes, int n_in,
                              void* d_out, int out_size, void* d_ws, size_t ws_size,
                              hipStream_t stream) {
    const float* x    = (const float*)d_in[0];
    const float* hid  = (const float*)d_in[1];
    const float* wxr  = (const float*)d_in[2];
    const float* whr  = (const float*)d_in[3];
    const float* br   = (const float*)d_in[4];
    const float* wxz  = (const float*)d_in[5];
    const float* whz  = (const float*)d_in[6];
    const float* bz   = (const float*)d_in[7];
    const float* wxh  = (const float*)d_in[8];
    const float* whh  = (const float*)d_in[9];
    const float* bh   = (const float*)d_in[10];
    const float* wc   = (const float*)d_in[11];
    const float* bc   = (const float*)d_in[12];
    const float* what = (const float*)d_in[13];
    const float* bhat = (const float*)d_in[14];
    float* out = (float*)d_out;
    (void)in_sizes; (void)n_in; (void)out_size; (void)ws_size;

    char* base = (char*)d_ws; size_t off = 0;
    auto take = [&](size_t bytes) -> void* {
        void* r = base + off;
        off = (off + bytes + 255) & ~(size_t)255;
        return r;
    };
    bf16_t* comb = (bf16_t*)take((size_t)BATCH * KCOMB * 2);  // 32 MB
    bf16_t* W4   = (bf16_t*)take((size_t)4 * HID * KCOMB * 2);// 16 MB
    bf16_t* Wxhb = (bf16_t*)take((size_t)HID * HID * 2);      //  2 MB
    bf16_t* Whhb = (bf16_t*)take((size_t)HID * HID * 2);      //  2 MB
    bf16_t* rh   = (bf16_t*)take((size_t)BATCH * HID * 2);    // 16 MB
    float*  cbuf = (float*)take((size_t)BATCH * HID * 4);     // 32 MB
    float*  hatb = (float*)take((size_t)BATCH * HID * 4);     // 32 MB
    float*  zbuf = (float*)take((size_t)BATCH * HID * 4);     // 32 MB
    float*  xhb  = (float*)take((size_t)BATCH * HID * 4);     // 32 MB (xh + hh)

    const int n_bh = (BATCH * HID + 255) / 256;

    k_build_combined<<<n_bh, 256, 0, stream>>>(x, hid, comb);
    k_pack_w4<<<(int)(((size_t)4 * HID * KCOMB + 255) / 256), 256, 0, stream>>>(
        wc, what, wxr, whr, wxz, whz, W4);
    k_cvt<<<(HID * HID + 255) / 256, 256, 0, stream>>>(wxh, Wxhb, (size_t)HID * HID);
    k_cvt<<<(HID * HID + 255) / 256, 256, 0, stream>>>(whh, Whhb, (size_t)HID * HID);

    // GEMM1: [8192,2048] x [4096,2048]^T, fused gate/attention epilogue.
    EpiFused1 e1{cbuf, hatb, rh, zbuf, bc, bhat, br, bz, hid};
    dim3 g1(4 * HID / 128, BATCH / 128);
    gemm_bf16_wmma<EpiFused1><<<g1, 256, 0, stream>>>(comb, KCOMB, W4, KCOMB, e1);

    softmax_rows<<<BATCH, 256, 0, stream>>>(cbuf);

    // GEMM2: x-part of combined (lda=2048, K=1024) @ Wxh^T + bias_h -> xhb
    EpiBias e2{xhb, bh};
    dim3 g2(HID / 128, BATCH / 128);
    gemm_bf16_wmma<EpiBias><<<g2, 256, 0, stream>>>(comb, KCOMB, Wxhb, HID, e2);

    // GEMM3: rh @ Whh^T accumulated onto xhb
    EpiAcc e3{xhb};
    gemm_bf16_wmma<EpiAcc><<<g2, 256, 0, stream>>>(rh, HID, Whhb, HID, e3);

    k_final<<<n_bh, 256, 0, stream>>>(zbuf, xhb, hid, cbuf, hatb, out);
}
